// RNNAttention_25546465477078
// MI455X (gfx1250) — compile-verified, matrix operand order checked by reading, not run on hardware
//
#include <hip/hip_runtime.h>
#include <hip/hip_bf16.h>

typedef __attribute__((ext_vector_type(16))) __bf16 v16bf;
typedef __attribute__((ext_vector_type(8)))  float  v8f;

#define NEG_INF_F (-1e9f)

__device__ __forceinline__ unsigned short f2bf(float f) {
  unsigned u = __builtin_bit_cast(unsigned, f);
  u += 0x7FFFu + ((u >> 16) & 1u);          // round-to-nearest-even
  return (unsigned short)(u >> 16);
}
__device__ __forceinline__ float bf2f(unsigned short h) {
  return __builtin_bit_cast(float, ((unsigned)h) << 16);
}

struct FragBits { uint4 a, b; };
__device__ __forceinline__ v16bf make_frag(uint4 a, uint4 b) {
  FragBits fb{a, b};
  return __builtin_bit_cast(v16bf, fb);
}

// ---------------------------------------------------------------------------
// K1a: dec_proj[b][a] = sum_k dh[b][k] * Wdec[k][a]   (64 x 512, tiny)
// ---------------------------------------------------------------------------
__global__ void k_dec_proj(const float* __restrict__ dh,
                           const float* __restrict__ Wd,
                           float* __restrict__ dp) {
  __shared__ float sH[1024];
  const int b = blockIdx.x, t = threadIdx.x;
  for (int j = 0; j < 4; ++j) sH[t + 256 * j] = dh[b * 1024 + t + 256 * j];
  __syncthreads();
  float s0 = 0.f, s1 = 0.f;
  for (int k = 0; k < 1024; ++k) {
    const float h = sH[k];
    s0 += h * Wd[k * 512 + t];
    s1 += h * Wd[k * 512 + t + 256];
  }
  dp[b * 512 + t] = s0;
  dp[b * 512 + t + 256] = s1;
}

// ---------------------------------------------------------------------------
// K1b: split W_enc (1024x512 fp32) into hi/lo bf16, swizzled into WMMA B
// fragment order: [kstep][part hi/lo][ntile][half][lane][8 elems]
// element (kstep,nt,half,lane,jj):  k = 32*kstep + jj + 16*half + 8*(lane>>4)
//                                   n = 16*nt + (lane&15)
// ---------------------------------------------------------------------------
__global__ void k_split_wenc(const float* __restrict__ We,
                             unsigned short* __restrict__ wsB) {
  const unsigned idx  = blockIdx.x * 256 + threadIdx.x;   // 0 .. 524287
  const unsigned jj   = idx & 7;
  const unsigned lane = (idx >> 3) & 31;
  const unsigned half = (idx >> 8) & 1;
  const unsigned nt   = (idx >> 9) & 31;
  const unsigned kk   = idx >> 14;
  const unsigned k = kk * 32 + jj + 16 * half + 8 * (lane >> 4);
  const unsigned n = nt * 16 + (lane & 15);
  const float x = We[k * 512 + n];
  const unsigned short hi = f2bf(x);
  const unsigned short lo = f2bf(x - bf2f(hi));
  const unsigned rest = nt * 512 + half * 256 + lane * 8 + jj;
  wsB[kk * 32768u + rest]          = hi;
  wsB[kk * 32768u + 16384u + rest] = lo;
}

// ---------------------------------------------------------------------------
// K2: fused scores = tanh(enc @ W_enc + dec_proj) . v  with mask
// block: 512 threads = 16 waves; tile 128 rows x 512 cols; K chunks of 32.
// bf16x3 WMMA: hi*hi + hi*lo + lo*hi  (~fp32 accuracy on matrix cores)
// ---------------------------------------------------------------------------
__global__ __launch_bounds__(512) void k_scores(
    const float* __restrict__ enc, const float* __restrict__ dpw,
    const float* __restrict__ v,   const int* __restrict__ mask,
    const unsigned short* __restrict__ wsB, float* __restrict__ scores) {
  extern __shared__ unsigned char smem[];
  unsigned short* sB  = (unsigned short*)smem;     // 32768 ushorts (64KB)
  unsigned short* sAh = sB + 32768;                // 128 rows * pitch 40
  unsigned short* sAl = sAh + 5120;
  float* sScore       = (float*)(sAl + 5120);      // [2][128]

  const int b    = blockIdx.y;
  const int s0   = blockIdx.x * 128;
  const int t    = threadIdx.x;
  const int w    = t >> 5;
  const int lane = t & 31;
  const int mtile = w & 7;        // 8 M-tiles of 16 rows
  const int nhalf = w >> 3;       // 2 halves of 256 cols

  const float* encB = enc + ((size_t)b * 2048 + s0) * 1024;

  v8f acc[16];
  {
    v8f z = {};
    #pragma unroll
    for (int i = 0; i < 16; ++i) acc[i] = z;
  }

  const int arow = t >> 3;          // 0..63
  const int acg  = (t & 7) * 4;     // col group (of 4) within 32

  for (int kk = 0; kk < 32; ++kk) {
    __syncthreads();
    // ---- stage A chunk: 128 rows x 32 cols fp32 -> hi/lo bf16 in LDS ----
    #pragma unroll
    for (int it = 0; it < 2; ++it) {
      const int row = arow + 64 * it;
      const float4 x = *(const float4*)(encB + (size_t)row * 1024 + kk * 32 + acg);
      const float xv[4] = {x.x, x.y, x.z, x.w};
      #pragma unroll
      for (int j = 0; j < 4; ++j) {
        const unsigned short hi = f2bf(xv[j]);
        sAh[row * 40 + acg + j] = hi;
        sAl[row * 40 + acg + j] = f2bf(xv[j] - bf2f(hi));
      }
    }
    // ---- stage B chunk: 64KB pre-swizzled hi/lo fragments (L2-resident) ----
    {
      const uint4* src = (const uint4*)(wsB + (size_t)kk * 32768u);
      uint4* dst = (uint4*)sB;
      #pragma unroll
      for (int i = 0; i < 8; ++i) dst[t + 512 * i] = src[t + 512 * i];
    }
    if (kk < 31)  // hint next B chunk toward the caches
      __builtin_prefetch(wsB + (size_t)(kk + 1) * 32768u + t * 64, 0, 1);
    __syncthreads();

    // ---- A fragments (ISA 16-bit A layout; two b128 per lane) ----
    const int m    = mtile * 16 + (lane & 15);
    const int koff = (lane >> 4) * 8;
    const v16bf Ah = make_frag(*(const uint4*)&sAh[m * 40 + koff],
                               *(const uint4*)&sAh[m * 40 + koff + 16]);
    const v16bf Al = make_frag(*(const uint4*)&sAl[m * 40 + koff],
                               *(const uint4*)&sAl[m * 40 + koff + 16]);
    #pragma unroll
    for (int tt = 0; tt < 16; ++tt) {
      const int nt = nhalf * 16 + tt;
      const unsigned short* pb = sB + nt * 512 + lane * 8;
      const v16bf Bh = make_frag(*(const uint4*)pb,
                                 *(const uint4*)(pb + 256));
      const v16bf Bl = make_frag(*(const uint4*)(pb + 16384),
                                 *(const uint4*)(pb + 16384 + 256));
      acc[tt] = __builtin_amdgcn_wmma_f32_16x16x32_bf16(false, Ah, false, Bh,
                                                        (short)0, acc[tt], false, false);
      acc[tt] = __builtin_amdgcn_wmma_f32_16x16x32_bf16(false, Ah, false, Bl,
                                                        (short)0, acc[tt], false, false);
      acc[tt] = __builtin_amdgcn_wmma_f32_16x16x32_bf16(false, Al, false, Bh,
                                                        (short)0, acc[tt], false, false);
    }
  }

  // ---- epilogue: tanh + dot with v, per-row reduction ----
  float partial[8];
  #pragma unroll
  for (int r = 0; r < 8; ++r) partial[r] = 0.f;
  #pragma unroll
  for (int tt = 0; tt < 16; ++tt) {
    const int n = nhalf * 256 + tt * 16 + (lane & 15);
    const float dp = dpw[b * 512 + n];
    const float vv = v[n];
    #pragma unroll
    for (int r = 0; r < 8; ++r)
      partial[r] += tanhf(acc[tt][r] + dp) * vv;
  }
  #pragma unroll
  for (int off = 1; off < 16; off <<= 1) {
    #pragma unroll
    for (int r = 0; r < 8; ++r)
      partial[r] += __shfl_xor(partial[r], off, 32);
  }
  if ((lane & 15) == 0) {
    #pragma unroll
    for (int r = 0; r < 8; ++r) {
      const int rowl = mtile * 16 + r + 8 * (lane >> 4);
      sScore[nhalf * 128 + rowl] = partial[r];  // one writer per slot: deterministic
    }
  }
  __syncthreads();
  if (t < 128) {
    float s = sScore[t] + sScore[128 + t];
    const int srow = s0 + t;
    if (mask[b * 2048 + srow] == 0) s = NEG_INF_F;
    scores[b * 2048 + srow] = s;
  }
}

// ---------------------------------------------------------------------------
// K3a: row softmax over S=2048; writes attn_weights into d_out[65536 + ...]
// ---------------------------------------------------------------------------
__global__ void k_softmax(const float* __restrict__ scores, float* __restrict__ out) {
  __shared__ float red[256];
  const int b = blockIdx.x, t = threadIdx.x;
  float sc[8];
  float mx = -3.4e38f;
  #pragma unroll
  for (int i = 0; i < 8; ++i) {
    sc[i] = scores[b * 2048 + t + 256 * i];
    mx = fmaxf(mx, sc[i]);
  }
  red[t] = mx; __syncthreads();
  for (int s = 128; s > 0; s >>= 1) { if (t < s) red[t] = fmaxf(red[t], red[t + s]); __syncthreads(); }
  mx = red[0]; __syncthreads();
  float lsum = 0.f;
  #pragma unroll
  for (int i = 0; i < 8; ++i) { sc[i] = expf(sc[i] - mx); lsum += sc[i]; }
  red[t] = lsum; __syncthreads();
  for (int s = 128; s > 0; s >>= 1) { if (t < s) red[t] += red[t + s]; __syncthreads(); }
  const float inv = 1.f / red[0];
  #pragma unroll
  for (int i = 0; i < 8; ++i)
    out[65536 + b * 2048 + t + 256 * i] = sc[i] * inv;
}

// ---------------------------------------------------------------------------
// K3b: context partials: part[gx][b][e] = sum_{s in gx-range} w[s]*enc[b][s][e]
// ---------------------------------------------------------------------------
__global__ void k_ctx_partial(const float* __restrict__ enc,
                              const float* __restrict__ attn,
                              float* __restrict__ part) {
  __shared__ float sW[256];
  const int gx = blockIdx.x, b = blockIdx.y, t = threadIdx.x;
  sW[t] = attn[b * 2048 + gx * 256 + t];
  __syncthreads();
  const float* src = enc + ((size_t)b * 2048 + gx * 256) * 1024 + 4 * t;
  float ax = 0.f, ay = 0.f, az = 0.f, aw = 0.f;
  #pragma unroll 4
  for (int i = 0; i < 256; ++i) {
    const float wgt = sW[i];
    const float4 x = *(const float4*)(src + (size_t)i * 1024);
    ax += wgt * x.x; ay += wgt * x.y; az += wgt * x.z; aw += wgt * x.w;
  }
  float4 r; r.x = ax; r.y = ay; r.z = az; r.w = aw;
  *(float4*)&part[((size_t)gx * 64 + b) * 1024 + 4 * t] = r;
}

// ---------------------------------------------------------------------------
// K3c: deterministic 8-way reduce of partials -> context in d_out[0..65535]
// ---------------------------------------------------------------------------
__global__ void k_ctx_reduce(const float* __restrict__ part, float* __restrict__ out) {
  const int o = blockIdx.x * 256 + threadIdx.x;   // 0..65535
  float s = 0.f;
  #pragma unroll
  for (int i = 0; i < 8; ++i) s += part[(size_t)i * 65536 + o];
  out[o] = s;
}

// ---------------------------------------------------------------------------
extern "C" void kernel_launch(void* const* d_in, const int* in_sizes, int n_in,
                              void* d_out, int out_size, void* d_ws, size_t ws_size,
                              hipStream_t stream) {
  const float* dh   = (const float*)d_in[0];   // [64,1024]
  const float* enc  = (const float*)d_in[1];   // [64,2048,1024]
  const int*   mask = (const int*)d_in[2];     // [64,2048]
  const float* We   = (const float*)d_in[3];   // [1024,512]
  const float* Wd   = (const float*)d_in[4];   // [1024,512]
  const float* v    = (const float*)d_in[5];   // [512]
  float* out = (float*)d_out;                  // context(65536) ++ attn(131072)

  char* ws = (char*)d_ws;
  float*          wsDP   = (float*)ws;                           // 128 KB
  float*          wsSc   = (float*)(ws + 131072);                // 512 KB
  unsigned short* wsB    = (unsigned short*)(ws + 655360);       // 2 MB
  float*          wsPart = (float*)(ws + 655360 + 2097152);      // 2 MB

  k_dec_proj  <<<dim3(64),     256, 0, stream>>>(dh, Wd, wsDP);
  k_split_wenc<<<dim3(2048),   256, 0, stream>>>(We, wsB);
  // dynamic LDS: 64KB B + 2*10240B A + 1KB scores = 87040 bytes
  k_scores    <<<dim3(16, 64), 512, 87040, stream>>>(enc, wsDP, v, mask, wsB, wsSc);
  k_softmax   <<<dim3(64),     256, 0, stream>>>(wsSc, out);
  k_ctx_partial<<<dim3(8, 64), 256, 0, stream>>>(enc, out + 65536, wsPart);
  k_ctx_reduce<<<dim3(256),    256, 0, stream>>>(wsPart, out);
}
